// FuseAttentionMLP_34437047779722
// MI455X (gfx1250) — matrix operand order repaired and checked by reading, not compiled, not run on hardware
//
#include <hip/hip_runtime.h>
#include <hip/hip_bf16.h>

// ---------------------------------------------------------------------------
// Types for CDNA5 WMMA (wave32)
// ---------------------------------------------------------------------------
typedef __bf16 bf16_t;
typedef __attribute__((ext_vector_type(16))) __bf16       v16bf;
typedef __attribute__((ext_vector_type(8)))  float        v8f;
typedef __attribute__((ext_vector_type(4)))  unsigned int v4u;

#define NHEADS 32
#define NKVH   8
#define HDIM   128
#define SEQ    1024
#define BATCH  2
#define DMODEL 4096
#define INTER  14336
#define PN     34816            // proj width = 4096 + 1024 + 1024 + 14336*2
#define TOK    (BATCH * SEQ)    // 2048

// D = A(16x32 bf16) x B(32x16 bf16) + C(16x16 f32)
__device__ inline v8f wmma_bf16(v16bf a, v16bf b, v8f c) {
  return __builtin_amdgcn_wmma_f32_16x16x32_bf16(
      /*neg_a=*/false, a, /*neg_b=*/false, b,
      /*c_mod=*/(short)0, c, /*reuse_a=*/false, /*reuse_b=*/false);
}

// A-operand fragment (16x32, row-major source): lane half selects K-interleave.
// p points at element (row m, k = hl*8); reads k={hl*8..+7, 16+hl*8..+7}
__device__ inline v16bf fragA(const bf16_t* p) {
  union { v4u u[2]; v16bf v; } f;
  f.u[0] = *reinterpret_cast<const v4u*>(p);
  f.u[1] = *reinterpret_cast<const v4u*>(p + 16);
  return f.v;
}
// B-operand fragment (32x16 = W^T, from row-major W[n][k]): 16 contiguous k.
__device__ inline v16bf fragB(const bf16_t* p) {
  union { v4u u[2]; v16bf v; } f;
  f.u[0] = *reinterpret_cast<const v4u*>(p);
  f.u[1] = *reinterpret_cast<const v4u*>(p + 8);
  return f.v;
}

// CDNA5 async global->LDS copy (ASYNCcnt-tracked, no VGPR staging).
// the LDS address comes from the dst VGPR; generic shared pointers carry the LDS offset
// in their low 32 bits (aperture in [63:32]).
__device__ inline void async_ld_b128(unsigned int lds_off, const bf16_t* g) {
  asm volatile("global_load_async_to_lds_b128 %0, %1, off"
               :: "v"(lds_off), "v"((unsigned long long)(uintptr_t)g)
               : "memory");
}
__device__ inline void wait_async0() {
  asm volatile("s_wait_asynccnt 0x0" ::: "memory");
}

// ---------------------------------------------------------------------------
// fp32 -> bf16 convert (grid-stride)
// ---------------------------------------------------------------------------
__global__ void cvt_f32_bf16_kernel(const float* __restrict__ s,
                                    bf16_t* __restrict__ d, size_t n) {
  size_t i = (size_t)blockIdx.x * blockDim.x + threadIdx.x;
  size_t stride = (size_t)gridDim.x * blockDim.x;
  for (; i < n; i += stride) d[i] = (bf16_t)s[i];
}

// ---------------------------------------------------------------------------
// GEMM: Cout[M,N] = A[M,K](bf16) @ W[N,K](bf16)^T  (+ Cin)   fp32 out
// block: 256 threads = 8 waves; tile 128x256x32; wave tile 64x64 (4x4 WMMA)
// Staging uses GLOBAL_LOAD_ASYNC_TO_LDS_B128 (ASYNCcnt).
// ---------------------------------------------------------------------------
#define LDSP 40  // padded bf16 row stride for a 32-wide k-slice

__global__ __launch_bounds__(256) void gemm_bf16_kernel(
    const bf16_t* __restrict__ A, const bf16_t* __restrict__ W,
    const float* __restrict__ Cin, float* __restrict__ Cout,
    int M, int N, int K) {
  __shared__ __align__(16) bf16_t As[128 * LDSP];
  __shared__ __align__(16) bf16_t Ws[256 * LDSP];

  const int tid  = threadIdx.x;
  const int lane = tid & 31;
  const int wv   = tid >> 5;
  const int wm   = (wv >> 2) * 64;  // wave row offset in tile
  const int wn   = (wv & 3) * 64;   // wave col offset in tile
  const int l15  = lane & 15;
  const int hl   = lane >> 4;

  const int m0 = blockIdx.y * 128;
  const int n0 = blockIdx.x * 256;

  v8f acc[4][4] = {};

  // staging: As = 512 16B-chunks (2/thread), Ws = 1024 chunks (4/thread)
  // chunk c: row = c>>2, seg = (c&3)*8 bf16 elements
  unsigned int a_lds[2];
  const bf16_t* a_g[2];
#pragma unroll
  for (int j = 0; j < 2; ++j) {
    int c = tid * 2 + j, row = c >> 2, seg = (c & 3) * 8;
    a_lds[j] = (unsigned int)(uintptr_t)&As[row * LDSP + seg];
    a_g[j]   = A + (size_t)(m0 + row) * K + seg;
  }
  unsigned int w_lds[4];
  const bf16_t* w_g[4];
#pragma unroll
  for (int j = 0; j < 4; ++j) {
    int c = tid * 4 + j, row = c >> 2, seg = (c & 3) * 8;
    w_lds[j] = (unsigned int)(uintptr_t)&Ws[row * LDSP + seg];
    w_g[j]   = W + (size_t)(n0 + row) * K + seg;
  }

  for (int k0 = 0; k0 < K; k0 += 32) {
    __syncthreads();  // previous iteration's readers done
#pragma unroll
    for (int j = 0; j < 2; ++j) async_ld_b128(a_lds[j], a_g[j] + k0);
#pragma unroll
    for (int j = 0; j < 4; ++j) async_ld_b128(w_lds[j], w_g[j] + k0);
    wait_async0();
    __syncthreads();
    if (k0 + 32 < K) {  // CDNA5 global_prefetch_b8 for the next k-slice
      __builtin_prefetch(a_g[0] + k0 + 32, 0, 1);
      __builtin_prefetch(w_g[0] + k0 + 32, 0, 1);
    }
    v16bf af[4], bfr[4];
#pragma unroll
    for (int mi = 0; mi < 4; ++mi)
      af[mi] = fragB(&As[(wm + mi * 16 + l15) * LDSP + hl * 16]);
#pragma unroll
    for (int ni = 0; ni < 4; ++ni)
      bfr[ni] = fragB(&Ws[(wn + ni * 16 + l15) * LDSP + hl * 16]);
#pragma unroll
    for (int mi = 0; mi < 4; ++mi)
#pragma unroll
      for (int ni = 0; ni < 4; ++ni)
        acc[mi][ni] = wmma_bf16(af[mi], bfr[ni], acc[mi][ni]);
  }

#pragma unroll
  for (int mi = 0; mi < 4; ++mi)
#pragma unroll
    for (int ni = 0; ni < 4; ++ni) {
      const int row = m0 + wm + mi * 16 + hl * 8;
      const int col = n0 + wn + ni * 16 + l15;
#pragma unroll
      for (int r = 0; r < 8; ++r) {
        size_t off = (size_t)(row + r) * N + col;
        float v = acc[mi][ni][r];
        if (Cin) v += Cin[off];
        Cout[off] = v;
      }
    }
}

// ---------------------------------------------------------------------------
// RoPE on Q; writes bf16 q [B,NH,S,HD]
// ---------------------------------------------------------------------------
__global__ void rope_q_kernel(const float* __restrict__ proj,
                              const float* __restrict__ fr,
                              bf16_t* __restrict__ qo) {
  int idx = blockIdx.x * blockDim.x + threadIdx.x;  // TOK*NH*64
  int j = idx & 63; int tmp = idx >> 6;
  int h = tmp & 31; int t = tmp >> 5;
  int b = t >> 10, s = t & 1023;
  const float* pr = proj + (size_t)t * PN + h * HDIM + 2 * j;
  float v0 = pr[0], v1 = pr[1];
  float c = fr[(s * 64 + j) * 2], sn = fr[(s * 64 + j) * 2 + 1];
  size_t o = ((size_t)(b * NHEADS + h) * SEQ + s) * HDIM + 2 * j;
  qo[o]     = (bf16_t)(v0 * c - v1 * sn);
  qo[o + 1] = (bf16_t)(v1 * c + v0 * sn);
}

// RoPE on K; writes bf16 k [B,NKVH,S,HD]
__global__ void rope_k_kernel(const float* __restrict__ proj,
                              const float* __restrict__ fr,
                              bf16_t* __restrict__ ko) {
  int idx = blockIdx.x * blockDim.x + threadIdx.x;  // TOK*NKVH*64
  int j = idx & 63; int tmp = idx >> 6;
  int h = tmp & 7; int t = tmp >> 3;
  int b = t >> 10, s = t & 1023;
  const float* pr = proj + (size_t)t * PN + DMODEL + h * HDIM + 2 * j;
  float v0 = pr[0], v1 = pr[1];
  float c = fr[(s * 64 + j) * 2], sn = fr[(s * 64 + j) * 2 + 1];
  size_t o = ((size_t)(b * NKVH + h) * SEQ + s) * HDIM + 2 * j;
  ko[o]     = (bf16_t)(v0 * c - v1 * sn);
  ko[o + 1] = (bf16_t)(v1 * c + v0 * sn);
}

// V transpose: bf16 vt [B,NKVH,HD,S]
__global__ void v_tr_kernel(const float* __restrict__ proj,
                            bf16_t* __restrict__ vt) {
  int idx = blockIdx.x * blockDim.x + threadIdx.x;  // TOK*NKVH*HD
  int d = idx & 127; int tmp = idx >> 7;
  int h = tmp & 7; int t = tmp >> 3;
  int b = t >> 10, s = t & 1023;
  float v = proj[(size_t)t * PN + DMODEL + NKVH * HDIM + h * HDIM + d];
  vt[((size_t)(b * NKVH + h) * HDIM + d) * SEQ + s] = (bf16_t)v;
}

// SwiGLU: h = silu(g) * u, bf16 [TOK, INTER]
__global__ void swiglu_kernel(const float* __restrict__ proj,
                              bf16_t* __restrict__ hout) {
  size_t i = (size_t)blockIdx.x * blockDim.x + threadIdx.x;
  size_t stride = (size_t)gridDim.x * blockDim.x;
  const size_t n = (size_t)TOK * INTER;
  for (; i < n; i += stride) {
    size_t t = i / INTER, j = i - t * INTER;
    float g = proj[t * PN + DMODEL + 2 * NKVH * HDIM + INTER + j];
    float u = proj[t * PN + DMODEL + 2 * NKVH * HDIM + j];
    hout[i] = (bf16_t)(u * g / (1.0f + __expf(-g)));
  }
}

// ---------------------------------------------------------------------------
// Flash attention (causal, GQA). One wave = 16 queries; block = 8 waves.
// Scores computed TRANSPOSED (T = K x Q^T) so exp(P) feeds the PxV WMMA's
// B operand with only half-wave shuffles.  Q [B,NH,S,HD], K [B,NKVH,S,HD],
// Vt [B,NKVH,HD,S] all bf16;  out bf16 [B*S, NH*HD].
// ---------------------------------------------------------------------------
__global__ __launch_bounds__(256) void attn_kernel(
    const bf16_t* __restrict__ Q, const bf16_t* __restrict__ K,
    const bf16_t* __restrict__ Vt, bf16_t* __restrict__ O) {
  const int lane = threadIdx.x & 31;
  const int wv   = threadIdx.x >> 5;
  const int l15  = lane & 15;
  const int hl   = lane >> 4;

  const int b   = blockIdx.x >> 5;       // NH = 32
  const int h   = blockIdx.x & 31;
  const int kvh = h >> 2;                // NH/NKVH = 4
  const int qbase = blockIdx.y * 128 + wv * 16;
  const int q_idx = qbase + l15;

  const bf16_t* Qp = Q  + (size_t)(b * NHEADS + h)  * SEQ * HDIM;
  const bf16_t* Kp = K  + (size_t)(b * NKVH + kvh)  * SEQ * HDIM;
  const bf16_t* Vp = Vt + (size_t)(b * NKVH + kvh)  * HDIM * SEQ;

  // Q^T as B operand: column q = l15, 4 frags over HD (k = 32 each)
  v16bf qf[4];
#pragma unroll
  for (int kk = 0; kk < 4; ++kk)
    qf[kk] = fragB(Qp + (size_t)q_idx * HDIM + kk * 32 + hl * 16);

  v8f oacc[8] = {};                       // O^T: rows = d, cols = q
  float mrun = -3.0e38f, lrun = 0.0f;
  const float scale = 0.08838834764831845f;  // 1/sqrt(128)

  for (int ks = 0; ks < qbase + 16; ks += 32) {
    v8f t0 = {}, t1 = {};                 // score tiles: keys ks..+15, +16..+31
#pragma unroll
    for (int kk = 0; kk < 4; ++kk) {
      v16bf ka = fragB(Kp + (size_t)(ks + l15)      * HDIM + kk * 32 + hl * 16);
      v16bf kb = fragB(Kp + (size_t)(ks + 16 + l15) * HDIM + kk * 32 + hl * 16);
      t0 = wmma_bf16(ka, qf[kk], t0);
      t1 = wmma_bf16(kb, qf[kk], t1);
    }
    // per-query (column) softmax stats
    float s0[8], s1[8];
    float mloc = -3.0e38f;
#pragma unroll
    for (int r = 0; r < 8; ++r) {
      int key0 = ks + hl * 8 + r;
      int key1 = key0 + 16;
      float a = (key0 <= q_idx) ? t0[r] * scale : -3.0e38f;
      float c = (key1 <= q_idx) ? t1[r] * scale : -3.0e38f;
      s0[r] = a; s1[r] = c;
      mloc = fmaxf(mloc, fmaxf(a, c));
    }
    mloc = fmaxf(mloc, __shfl_xor(mloc, 16, 32));
    float mnew = fmaxf(mrun, mloc);
    float rsum = 0.0f;
#pragma unroll
    for (int r = 0; r < 8; ++r) {
      s0[r] = __expf(s0[r] - mnew);
      s1[r] = __expf(s1[r] - mnew);
      rsum += s0[r] + s1[r];
    }
    rsum += __shfl_xor(rsum, 16, 32);
    float alpha = __expf(mrun - mnew);
    mrun = mnew;
    lrun = lrun * alpha + rsum;

    // assemble P^T as B operand (k rows 0..31, col q) via half-wave exchange
    float x0[8], x1[8];
#pragma unroll
    for (int r = 0; r < 8; ++r) {
      x0[r] = __shfl_xor(s0[r], 16, 32);
      x1[r] = __shfl_xor(s1[r], 16, 32);
    }
    v16bf pf;
#pragma unroll
    for (int i = 0; i < 8; ++i) {
      pf[i]     = (bf16_t)((hl == 0) ? s0[i] : x1[i]);
      pf[i + 8] = (bf16_t)((hl == 0) ? x0[i] : s1[i]);
    }
    // O^T += V^T x P^T  (A = V^T rows d, contraction over 32 keys)
#pragma unroll
    for (int t = 0; t < 8; ++t) {
      oacc[t] = oacc[t] * alpha;
      v16bf vf = fragB(Vp + (size_t)(t * 16 + l15) * SEQ + ks + hl * 16);
      oacc[t] = wmma_bf16(vf, pf, oacc[t]);
    }
  }

  const float inv = 1.0f / lrun;
  const size_t orow = ((size_t)b * SEQ + q_idx) * (NHEADS * HDIM) + (size_t)h * HDIM;
#pragma unroll
  for (int t = 0; t < 8; ++t)
#pragma unroll
    for (int r = 0; r < 8; ++r)
      O[orow + t * 16 + hl * 8 + r] = (bf16_t)(oacc[t][r] * inv);
}

// ---------------------------------------------------------------------------
// Launch: inputs = {x, freqs_cis, mask(unused), wqkv1, wo, w2}; out fp32.
// ---------------------------------------------------------------------------
extern "C" void kernel_launch(void* const* d_in, const int* in_sizes, int n_in,
                              void* d_out, int out_size, void* d_ws, size_t ws_size,
                              hipStream_t stream) {
  const float* x     = (const float*)d_in[0];
  const float* freqs = (const float*)d_in[1];
  const float* wqkv1 = (const float*)d_in[3];
  const float* wo    = (const float*)d_in[4];
  const float* w2    = (const float*)d_in[5];
  float* out = (float*)d_out;

  char* p = (char*)d_ws;
  bf16_t* wqkv_bf = (bf16_t*)p; p += (size_t)PN * DMODEL * 2;
  bf16_t* wo_bf   = (bf16_t*)p; p += (size_t)DMODEL * DMODEL * 2;
  bf16_t* w2_bf   = (bf16_t*)p; p += (size_t)DMODEL * INTER * 2;
  bf16_t* x_bf    = (bf16_t*)p; p += (size_t)TOK * DMODEL * 2;
  bf16_t* q_bf    = (bf16_t*)p; p += (size_t)BATCH * NHEADS * SEQ * HDIM * 2;
  bf16_t* k_bf    = (bf16_t*)p; p += (size_t)BATCH * NKVH * SEQ * HDIM * 2;
  bf16_t* vt_bf   = (bf16_t*)p; p += (size_t)BATCH * NKVH * SEQ * HDIM * 2;
  bf16_t* attn_bf = (bf16_t*)p; p += (size_t)TOK * DMODEL * 2;
  bf16_t* h_bf    = (bf16_t*)p; p += (size_t)TOK * INTER * 2;
  float*  proj    = (float*)p;  p += (size_t)TOK * PN * 4;
  float*  yf      = (float*)p;  p += (size_t)TOK * DMODEL * 4;

  // 1) converts to bf16
  cvt_f32_bf16_kernel<<<8192, 256, 0, stream>>>(wqkv1, wqkv_bf, (size_t)PN * DMODEL);
  cvt_f32_bf16_kernel<<<8192, 256, 0, stream>>>(wo,    wo_bf,   (size_t)DMODEL * DMODEL);
  cvt_f32_bf16_kernel<<<8192, 256, 0, stream>>>(w2,    w2_bf,   (size_t)DMODEL * INTER);
  cvt_f32_bf16_kernel<<<8192, 256, 0, stream>>>(x,     x_bf,    (size_t)TOK * DMODEL);

  // 2) fused QKV + gate/up projection: proj = x @ wqkv1^T
  gemm_bf16_kernel<<<dim3(PN / 256, TOK / 128), 256, 0, stream>>>(
      x_bf, wqkv_bf, nullptr, proj, TOK, PN, DMODEL);

  // 3) RoPE + V transpose
  rope_q_kernel<<<(TOK * NHEADS * 64) / 256, 256, 0, stream>>>(proj, freqs, q_bf);
  rope_k_kernel<<<(TOK * NKVH * 64) / 256, 256, 0, stream>>>(proj, freqs, k_bf);
  v_tr_kernel<<<(TOK * NKVH * HDIM) / 256, 256, 0, stream>>>(proj, vt_bf);

  // 4) causal GQA flash attention
  attn_kernel<<<dim3(BATCH * NHEADS, SEQ / 128), 256, 0, stream>>>(
      q_bf, k_bf, vt_bf, attn_bf);

  // 5) output projection: y = attn @ wo^T
  gemm_bf16_kernel<<<dim3(DMODEL / 256, TOK / 128), 256, 0, stream>>>(
      attn_bf, wo_bf, nullptr, yf, TOK, DMODEL, DMODEL);

  // 6) SwiGLU
  swiglu_kernel<<<8192, 256, 0, stream>>>(proj, h_bf);

  // 7) out = h @ w2^T + y
  gemm_bf16_kernel<<<dim3(DMODEL / 256, TOK / 128), 256, 0, stream>>>(
      h_bf, w2_bf, yf, out, TOK, DMODEL, INTER);
}
